// VS_DiT_Block_85409719649071
// MI455X (gfx1250) — matrix operand order repaired and checked by reading, not compiled, not run on hardware
//
#include <hip/hip_runtime.h>
#include <hip/hip_bf16.h>
#include <math.h>

// ---- CDNA5 WMMA types -------------------------------------------------------
typedef __attribute__((ext_vector_type(16))) __bf16 v16bf;
typedef __attribute__((ext_vector_type(8)))  __bf16 v8bf;
typedef __attribute__((ext_vector_type(4)))  __bf16 v4bf;
typedef __attribute__((ext_vector_type(8)))  float  v8f;

#define BM 128
#define BN 128
#define BK 32
#define KPAD 8                 // 8 bf16 = 16B: keeps 16B alignment, kills bank conflicts
#define LDT (BK + KPAD)        // LDS row stride in elements (80 bytes)

__device__ __forceinline__ float silu_f(float x) { return x / (1.f + __expf(-x)); }
__device__ __forceinline__ float gelu_f(float x) { return 0.5f * x * (1.f + erff(x * 0.70710678118654752f)); }

// CDNA5 async global->LDS copy (16B per lane), tracked by ASYNCcnt.
__device__ __forceinline__ void async_ld_b128(unsigned lds_addr, const void* gaddr) {
  asm volatile("global_load_async_to_lds_b128 %0, %1, off" :: "v"(lds_addr), "v"(gaddr) : "memory");
}
__device__ __forceinline__ void wait_async0() {
  asm volatile("s_wait_asynccnt 0x0" ::: "memory");
}

// =============================================================================
// Tiled bf16 WMMA GEMM:  C[M,N] = epi( A[M,K] @ W[N,K]^T + bias[N] )
// A, W already bf16 in HBM. Double-buffered LDS staged via async-to-LDS DMA.
// EPI: 0=none, 1=exact gelu.  OBF16: bf16 output, else fp32.
// =============================================================================
template <int EPI, bool OBF16>
__global__ __launch_bounds__(256)
void gemm_wmma_kernel(const __bf16* __restrict__ A, const __bf16* __restrict__ W,
                      const float* __restrict__ bias, void* __restrict__ Cout,
                      int M, int N, int K)
{
  __shared__ __bf16 Ash[2][BM][LDT];
  __shared__ __bf16 Wsh[2][BM][LDT];

  const int tid  = threadIdx.x;
  const int lane = tid & 31;
  const int wave = tid >> 5;
  const int bm   = blockIdx.y * BM;
  const int bn   = blockIdx.x * BN;
  const int wm   = (wave & 1) * 64;    // 2 waves along M
  const int wn   = (wave >> 1) * 32;   // 4 waves along N

  v8f acc[4][2];
#pragma unroll
  for (int i = 0; i < 4; ++i)
#pragma unroll
    for (int j = 0; j < 2; ++j)
      acc[i][j] = (v8f){0.f, 0.f, 0.f, 0.f, 0.f, 0.f, 0.f, 0.f};

  // Staging: 512 chunks of 16B per tile; each thread owns 2 A-chunks + 2 W-chunks.
  const int c0   = tid * 2;
  const int crow = c0 >> 2;            // 0..127
  const int ccol = (c0 & 3) * 8;       // 0,16 (elements)
  const __bf16* ag = A + (size_t)(bm + crow) * K + ccol;
  const __bf16* wg = W + (size_t)(bn + crow) * K + ccol;

  auto stage = [&](int buf, int k0) {
    const unsigned la = (unsigned)(uintptr_t)&Ash[buf][crow][ccol];
    const unsigned lw = (unsigned)(uintptr_t)&Wsh[buf][crow][ccol];
    async_ld_b128(la,      ag + k0);
    async_ld_b128(la + 16, ag + k0 + 8);
    async_ld_b128(lw,      wg + k0);
    async_ld_b128(lw + 16, wg + k0 + 8);
  };

  stage(0, 0);
  wait_async0();
  __syncthreads();

  int buf = 0;
  for (int k0 = 0; k0 < K; k0 += BK) {
    if (k0 + BK < K) stage(buf ^ 1, k0 + BK);   // overlap DMA with WMMA

    // ---- fragment assembly per ISA 16-bit layouts ----
    // A 16x32: lanes 0-15 -> M=lane, K={0..7}U{16..23}; lanes 16-31 -> +8 K offset
    v16bf afrag[4], bfrag[2];
    const int arow = lane & 15;
    const int kA   = (lane < 16) ? 0 : 8;
#pragma unroll
    for (int mi = 0; mi < 4; ++mi) {
      const __bf16* ap = &Ash[buf][wm + mi * 16 + arow][0];
#pragma unroll
      for (int v = 0; v < 4; ++v) {
        afrag[mi][2 * v]         = ap[kA + 2 * v];
        afrag[mi][2 * v + 1]     = ap[kA + 2 * v + 1];
        afrag[mi][8 + 2 * v]     = ap[kA + 16 + 2 * v];
        afrag[mi][8 + 2 * v + 1] = ap[kA + 16 + 2 * v + 1];
      }
    }
    // B 32x16: lanes 0-15 hold K={0..7}U{16..23}; lanes 16-31 -> +8 K offset (same as A)
    const int bcol = lane & 15;
    const int kB   = (lane < 16) ? 0 : 8;
#pragma unroll
    for (int ni = 0; ni < 2; ++ni) {
      const __bf16* bp = &Wsh[buf][wn + ni * 16 + bcol][0];
#pragma unroll
      for (int v = 0; v < 8; ++v) {
        bfrag[ni][2 * v]     = bp[kB + 2 * v + (v >= 4 ? 8 : 0)];
        bfrag[ni][2 * v + 1] = bp[kB + 2 * v + 1 + (v >= 4 ? 8 : 0)];
      }
    }
#pragma unroll
    for (int mi = 0; mi < 4; ++mi)
#pragma unroll
      for (int ni = 0; ni < 2; ++ni)
        acc[mi][ni] = __builtin_amdgcn_wmma_f32_16x16x32_bf16(
            false, afrag[mi], false, bfrag[ni], (short)0, acc[mi][ni], false, false);

    wait_async0();        // next tile landed (this wave's DMA)
    __syncthreads();      // all waves done reading buf + all DMAs visible
    buf ^= 1;
  }

  // ---- epilogue: C/D layout: VGPR v -> M = v (lanes 0-15) / v+8 (lanes 16-31)
  const int cn    = lane & 15;
  const int mhalf = (lane < 16) ? 0 : 8;
#pragma unroll
  for (int ni = 0; ni < 2; ++ni) {
    const int gn   = bn + wn + ni * 16 + cn;
    const float bv = bias[gn];
#pragma unroll
    for (int mi = 0; mi < 4; ++mi) {
#pragma unroll
      for (int v = 0; v < 8; ++v) {
        const int gm = bm + wm + mi * 16 + mhalf + v;
        float x = acc[mi][ni][v] + bv;
        if (EPI == 1) x = gelu_f(x);
        if (OBF16) ((__bf16*)Cout)[(size_t)gm * N + gn] = (__bf16)x;
        else       ((float*)Cout)[(size_t)gm * N + gn]  = x;
      }
    }
  }
}

// ---- one-time fp32 -> bf16 conversion (optionally fused silu), 8 elems/thread
template <int SILU>
__global__ __launch_bounds__(256)
void cvt_kernel(const float* __restrict__ in, __bf16* __restrict__ out, long n)
{
  const long i = ((long)blockIdx.x * 256 + threadIdx.x) * 8;
  if (i >= n) return;
  float4 a = *(const float4*)(in + i);
  float4 b = *(const float4*)(in + i + 4);
  if (SILU) {
    a.x = silu_f(a.x); a.y = silu_f(a.y); a.z = silu_f(a.z); a.w = silu_f(a.w);
    b.x = silu_f(b.x); b.y = silu_f(b.y); b.z = silu_f(b.z); b.w = silu_f(b.w);
  }
  v8bf o;
  o[0] = (__bf16)a.x; o[1] = (__bf16)a.y; o[2] = (__bf16)a.z; o[3] = (__bf16)a.w;
  o[4] = (__bf16)b.x; o[5] = (__bf16)b.y; o[6] = (__bf16)b.z; o[7] = (__bf16)b.w;
  *(v8bf*)(out + i) = o;
}

// small fp32 copy (bias concat), one block of 256, 4 elems/thread
__global__ void copy_f32_kernel(const float* __restrict__ in, float* __restrict__ out)
{
  const int c = threadIdx.x * 4;
  *(float4*)(out + c) = *(const float4*)(in + c);
}

// =============================================================================
// Row LayerNorm + adaLN modulation -> bf16:  m[b,:] = ln(x[b,:])*(1+g)+be
// =============================================================================
__global__ __launch_bounds__(256)
void modln_kernel(const float* __restrict__ xin, const float* __restrict__ mod,
                  int gofs, int beofs, __bf16* __restrict__ mout)
{
  const int b   = blockIdx.x;
  const int tid = threadIdx.x;
  __shared__ float red[256];

  const float4 xv = ((const float4*)(xin + (size_t)b * 1024))[tid];
  red[tid] = xv.x + xv.y + xv.z + xv.w; __syncthreads();
  for (int off = 128; off > 0; off >>= 1) { if (tid < off) red[tid] += red[tid + off]; __syncthreads(); }
  const float mean = red[0] * (1.f / 1024.f);
  __syncthreads();

  const float d0 = xv.x - mean, d1 = xv.y - mean, d2 = xv.z - mean, d3 = xv.w - mean;
  red[tid] = d0 * d0 + d1 * d1 + d2 * d2 + d3 * d3; __syncthreads();
  for (int off = 128; off > 0; off >>= 1) { if (tid < off) red[tid] += red[tid + off]; __syncthreads(); }
  const float rstd = rsqrtf(red[0] * (1.f / 1024.f) + 1e-6f);

  const float4 g  = *(const float4*)(mod + (size_t)b * 9216 + gofs  + tid * 4);
  const float4 be = *(const float4*)(mod + (size_t)b * 9216 + beofs + tid * 4);
  v4bf o;
  o[0] = (__bf16)(d0 * rstd * (1.f + g.x) + be.x);
  o[1] = (__bf16)(d1 * rstd * (1.f + g.y) + be.y);
  o[2] = (__bf16)(d2 * rstd * (1.f + g.z) + be.z);
  o[3] = (__bf16)(d3 * rstd * (1.f + g.w) + be.w);
  *(v4bf*)(mout + (size_t)b * 1024 + tid * 4) = o;
}

// out[b,:] = base[b,:] + a[b,:] * val[b,:]
__global__ __launch_bounds__(256)
void residual_kernel(const float* __restrict__ base, const float* __restrict__ mod,
                     int aofs, const float* __restrict__ val, float* __restrict__ out)
{
  const int b = blockIdx.x, c = threadIdx.x * 4;
  const float4 bs = *(const float4*)(base + (size_t)b * 1024 + c);
  const float4 av = *(const float4*)(mod  + (size_t)b * 9216 + aofs + c);
  const float4 vv = *(const float4*)(val  + (size_t)b * 1024 + c);
  float4 o;
  o.x = bs.x + av.x * vv.x; o.y = bs.y + av.y * vv.y;
  o.z = bs.z + av.z * vv.z; o.w = bs.w + av.w * vv.w;
  *(float4*)(out + (size_t)b * 1024 + c) = o;
}

// =============================================================================
// Cross-attention, Lq=1. kv packed: row*2048, k at +h*64, v at +1024+h*64.
// One block per batch, 8 waves x 2 heads. Writes bf16.
// =============================================================================
__global__ __launch_bounds__(256)
void cross_attn_kernel(const __bf16* __restrict__ qb, const __bf16* __restrict__ kv,
                       __bf16* __restrict__ out)
{
  const int b    = blockIdx.x;
  const int wave = threadIdx.x >> 5;
  const int lane = threadIdx.x & 31;
  __shared__ float sc[8][80];

  for (int h = wave; h < 16; h += 8) {
    const float q0 = (float)qb[(size_t)b * 1024 + h * 64 + lane];
    const float q1 = (float)qb[(size_t)b * 1024 + h * 64 + 32 + lane];
    for (int l = 0; l < 77; ++l) {
      const __bf16* kp = kv + ((size_t)b * 77 + l) * 2048 + h * 64;
      float p = q0 * (float)kp[lane] + q1 * (float)kp[32 + lane];
      for (int off = 16; off > 0; off >>= 1) p += __shfl_xor(p, off, 32);
      if (lane == 0) sc[wave][l] = p * 0.125f;   // 1/sqrt(64)
    }
    __syncthreads();
    const float s0 = sc[wave][lane];
    const float s1 = sc[wave][lane + 32];
    const float s2 = (lane + 64 < 77) ? sc[wave][lane + 64] : -1e30f;
    float mx = fmaxf(s0, fmaxf(s1, s2));
    for (int off = 16; off > 0; off >>= 1) mx = fmaxf(mx, __shfl_xor(mx, off, 32));
    const float e0 = __expf(s0 - mx), e1 = __expf(s1 - mx);
    const float e2 = (lane + 64 < 77) ? __expf(s2 - mx) : 0.f;
    float sum = e0 + e1 + e2;
    for (int off = 16; off > 0; off >>= 1) sum += __shfl_xor(sum, off, 32);
    const float inv = 1.f / sum;
    sc[wave][lane] = e0 * inv;
    sc[wave][lane + 32] = e1 * inv;
    if (lane + 64 < 77) sc[wave][lane + 64] = e2 * inv;
    __syncthreads();
    float o0 = 0.f, o1 = 0.f;
    for (int l = 0; l < 77; ++l) {
      const float p = sc[wave][l];
      const __bf16* vp = kv + ((size_t)b * 77 + l) * 2048 + 1024 + h * 64;
      o0 += p * (float)vp[lane];
      o1 += p * (float)vp[32 + lane];
    }
    out[(size_t)b * 1024 + h * 64 + lane]      = (__bf16)o0;
    out[(size_t)b * 1024 + h * 64 + 32 + lane] = (__bf16)o1;
    __syncthreads();
  }
}

// =============================================================================
extern "C" void kernel_launch(void* const* d_in, const int* in_sizes, int n_in,
                              void* d_out, int out_size, void* d_ws, size_t ws_size,
                              hipStream_t stream)
{
  (void)in_sizes; (void)n_in; (void)out_size; (void)ws_size;
  const float* x      = (const float*)d_in[0];
  const float* t_emb  = (const float*)d_in[1];
  const float* ctx    = (const float*)d_in[2];
  const float* w_mod  = (const float*)d_in[4];
  const float* b_mod  = (const float*)d_in[5];
  const float* sa_wv  = (const float*)d_in[10];
  const float* sa_bv  = (const float*)d_in[11];
  const float* sa_wo  = (const float*)d_in[12];
  const float* sa_bo  = (const float*)d_in[13];
  const float* ca_wq  = (const float*)d_in[14];
  const float* ca_bq  = (const float*)d_in[15];
  const float* ca_wk  = (const float*)d_in[16];
  const float* ca_bk  = (const float*)d_in[17];
  const float* ca_wv  = (const float*)d_in[18];
  const float* ca_bv  = (const float*)d_in[19];
  const float* ca_wo  = (const float*)d_in[20];
  const float* ca_bo  = (const float*)d_in[21];
  const float* w1 = (const float*)d_in[22]; const float* b1 = (const float*)d_in[23];
  const float* w2 = (const float*)d_in[24]; const float* b2 = (const float*)d_in[25];
  const float* w3 = (const float*)d_in[26]; const float* b3 = (const float*)d_in[27];

  char* ws = (char*)d_ws;
  size_t ofs = 0;
  auto alloc = [&](size_t bytes) { char* p = ws + ofs; ofs += (bytes + 255) & ~(size_t)255; return p; };
  // bf16 weight / input copies (converted once)
  __bf16* tsil  = (__bf16*)alloc((size_t)1024 * 1024 * 2);       // silu(t_emb)
  __bf16* wmodb = (__bf16*)alloc((size_t)9216 * 1024 * 2);
  __bf16* sawvb = (__bf16*)alloc((size_t)1024 * 1024 * 2);
  __bf16* sawob = (__bf16*)alloc((size_t)1024 * 1024 * 2);
  __bf16* caqb  = (__bf16*)alloc((size_t)1024 * 1024 * 2);
  __bf16* caob  = (__bf16*)alloc((size_t)1024 * 1024 * 2);
  __bf16* wkvb  = (__bf16*)alloc((size_t)2048 * 1024 * 2);       // [wk ; wv]
  __bf16* w1b   = (__bf16*)alloc((size_t)4096 * 1024 * 2);
  __bf16* w2b   = (__bf16*)alloc((size_t)4096 * 4096 * 2);
  __bf16* w3b   = (__bf16*)alloc((size_t)1024 * 4096 * 2);
  __bf16* ctxb  = (__bf16*)alloc((size_t)78848 * 1024 * 2);
  float*  bkv   = (float*)alloc((size_t)2048 * 4);               // [bk ; bv]
  // activations / intermediates
  float*  mod   = (float*)alloc((size_t)1024 * 9216 * 4);
  float*  xq    = (float*)alloc((size_t)1024 * 1024 * 4);
  float*  t2    = (float*)alloc((size_t)1024 * 1024 * 4);
  __bf16* mb    = (__bf16*)alloc((size_t)1024 * 1024 * 2);
  __bf16* t1b   = (__bf16*)alloc((size_t)1024 * 1024 * 2);       // sa-v, later ca-q
  __bf16* cattb = (__bf16*)alloc((size_t)1024 * 1024 * 2);
  __bf16* h1b   = (__bf16*)alloc((size_t)1024 * 4096 * 2);
  __bf16* h2b   = (__bf16*)alloc((size_t)1024 * 4096 * 2);
  __bf16* kvb   = (__bf16*)alloc((size_t)78848 * 2048 * 2);

  const dim3 blk(256);
  auto cgrid = [](long n) { return dim3((unsigned)(n / 2048)); };
  auto ggrid = [](int M, int N) { return dim3((unsigned)(N / BN), (unsigned)(M / BM)); };

  // ---- one-time bf16 conversions ----
  cvt_kernel<1><<<cgrid(1024L * 1024), blk, 0, stream>>>(t_emb, tsil, 1024L * 1024);
  cvt_kernel<0><<<cgrid(9216L * 1024), blk, 0, stream>>>(w_mod, wmodb, 9216L * 1024);
  cvt_kernel<0><<<cgrid(1024L * 1024), blk, 0, stream>>>(sa_wv, sawvb, 1024L * 1024);
  cvt_kernel<0><<<cgrid(1024L * 1024), blk, 0, stream>>>(sa_wo, sawob, 1024L * 1024);
  cvt_kernel<0><<<cgrid(1024L * 1024), blk, 0, stream>>>(ca_wq, caqb,  1024L * 1024);
  cvt_kernel<0><<<cgrid(1024L * 1024), blk, 0, stream>>>(ca_wo, caob,  1024L * 1024);
  cvt_kernel<0><<<cgrid(1024L * 1024), blk, 0, stream>>>(ca_wk, wkvb,                1024L * 1024);
  cvt_kernel<0><<<cgrid(1024L * 1024), blk, 0, stream>>>(ca_wv, wkvb + 1024L * 1024, 1024L * 1024);
  cvt_kernel<0><<<cgrid(4096L * 1024), blk, 0, stream>>>(w1, w1b, 4096L * 1024);
  cvt_kernel<0><<<cgrid(4096L * 4096), blk, 0, stream>>>(w2, w2b, 4096L * 4096);
  cvt_kernel<0><<<cgrid(4096L * 1024), blk, 0, stream>>>(w3, w3b, 4096L * 1024);
  cvt_kernel<0><<<cgrid(78848L * 1024), blk, 0, stream>>>(ctx, ctxb, 78848L * 1024);
  copy_f32_kernel<<<1, blk, 0, stream>>>(ca_bk, bkv);
  copy_f32_kernel<<<1, blk, 0, stream>>>(ca_bv, bkv + 1024);

  // 1) mod = silu(t_emb) @ w_mod^T + b_mod
  gemm_wmma_kernel<0, false><<<ggrid(1024, 9216), blk, 0, stream>>>(tsil, wmodb, b_mod, mod, 1024, 9216, 1024);
  // 2) self-attention (Lq=Lk=1 -> softmax==1): sa = (m1@wv^T+bv)@wo^T+bo
  modln_kernel<<<1024, blk, 0, stream>>>(x, mod, 0, 1024, mb);
  gemm_wmma_kernel<0, true ><<<ggrid(1024, 1024), blk, 0, stream>>>(mb, sawvb, sa_bv, t1b, 1024, 1024, 1024);
  gemm_wmma_kernel<0, false><<<ggrid(1024, 1024), blk, 0, stream>>>(t1b, sawob, sa_bo, t2, 1024, 1024, 1024);
  residual_kernel<<<1024, blk, 0, stream>>>(x, mod, 2048, t2, xq);
  // 3) cross-attention: fused k+v projection (context read once)
  modln_kernel<<<1024, blk, 0, stream>>>(xq, mod, 3072, 4096, mb);
  gemm_wmma_kernel<0, true ><<<ggrid(1024, 1024), blk, 0, stream>>>(mb, caqb, ca_bq, t1b, 1024, 1024, 1024);
  gemm_wmma_kernel<0, true ><<<ggrid(78848, 2048), blk, 0, stream>>>(ctxb, wkvb, bkv, kvb, 78848, 2048, 1024);
  cross_attn_kernel<<<1024, blk, 0, stream>>>(t1b, kvb, cattb);
  gemm_wmma_kernel<0, false><<<ggrid(1024, 1024), blk, 0, stream>>>(cattb, caob, ca_bo, t2, 1024, 1024, 1024);
  residual_kernel<<<1024, blk, 0, stream>>>(xq, mod, 5120, t2, xq);
  // 4) FFN with fused exact-gelu epilogues
  modln_kernel<<<1024, blk, 0, stream>>>(xq, mod, 6144, 7168, mb);
  gemm_wmma_kernel<1, true ><<<ggrid(1024, 4096), blk, 0, stream>>>(mb,  w1b, b1, h1b, 1024, 4096, 1024);
  gemm_wmma_kernel<1, true ><<<ggrid(1024, 4096), blk, 0, stream>>>(h1b, w2b, b2, h2b, 1024, 4096, 4096);
  gemm_wmma_kernel<0, false><<<ggrid(1024, 1024), blk, 0, stream>>>(h2b, w3b, b3, t2, 1024, 1024, 4096);
  // 5) out = xq + a3*ff
  residual_kernel<<<1024, blk, 0, stream>>>(xq, mod, 8192, t2, (float*)d_out);
}
